// LSTMFourierDecoder_50130858279029
// MI455X (gfx1250) — matrix operand order repaired and checked by reading, not compiled, not run on hardware
//
#include <hip/hip_runtime.h>
#include <hip/hip_bf16.h>

// Problem dimensions (fixed by the reference).
#define BDIM 64
#define DDIM 512
#define TDIM 1024
#define LDIM 512
#define BT   (BDIM * TDIM)   // 65536 rows (b,t)

#define BSTRIDE 40           // LDS panel row pitch in shorts: 80B = 20 banks, 16B aligned

typedef __attribute__((ext_vector_type(16))) __bf16        bf16x16;
typedef __attribute__((ext_vector_type(8)))  float         f32x8;
typedef __attribute__((ext_vector_type(4)))  float         f32x4;
typedef __attribute__((ext_vector_type(4)))  unsigned int  u32x4;

union Frag {
    bf16x16        v;
    u32x4          q[2];
    unsigned short us[16];
};

__device__ __forceinline__ unsigned short f2bf(float f) {
    union { float f; unsigned u; } c; c.f = f;
    unsigned r = c.u + 0x7FFFu + ((c.u >> 16) & 1u);   // round-to-nearest-even
    return (unsigned short)(r >> 16);
}
__device__ __forceinline__ float bf2f(unsigned short h) {
    union { unsigned u; float f; } c; c.u = ((unsigned)h) << 16;
    return c.f;
}

// Async 16B global->LDS copy (gfx1250 GLOBAL_LOAD_ASYNC_TO_LDS_B128, ASYNCcnt).
__device__ __forceinline__ void async_copy16(void* lds_ptr, const void* gptr) {
    unsigned lds_off = (unsigned)(reinterpret_cast<uintptr_t>(lds_ptr)); // addr[31:0] = LDS offset
    asm volatile("global_load_async_to_lds_b128 %0, %1, off"
                 :: "v"(lds_off), "v"(gptr) : "memory");
}
__device__ __forceinline__ void async_wait() {
    asm volatile("s_wait_asynccnt 0x0" ::: "memory");
}

// ---------------------------------------------------------------------------
// Kernel 1: convert the 7 [512x512] f32 weight matrices to bf16, TRANSPOSED
// (dst[mat][n][k] = src[mat][k][n]) so WMMA B-fragments are contiguous.
// Slots: 0=rff_w, 1..3 = W_i/f/g, 4..6 = R_i/f/g.
// ---------------------------------------------------------------------------
__global__ __launch_bounds__(256, 1) void wcvt_kernel(
    const float* __restrict__ w0, const float* __restrict__ w1,
    const float* __restrict__ w2, const float* __restrict__ w3,
    const float* __restrict__ w4, const float* __restrict__ w5,
    const float* __restrict__ w6, unsigned short* __restrict__ dst)
{
    int idx = blockIdx.x * 256 + threadIdx.x;       // < 7*262144
    int mat = idx >> 18;
    int rem = idx & 262143;
    int n   = rem >> 9;
    int k   = rem & 511;
    const float* src = (mat == 0) ? w0 : (mat == 1) ? w1 : (mat == 2) ? w2 :
                       (mat == 3) ? w3 : (mat == 4) ? w4 : (mat == 5) ? w5 : w6;
    dst[idx] = f2bf(src[k * 512 + n]);
}

// ---------------------------------------------------------------------------
// Kernel 2: transpose x [B][D][T] f32 -> xbf [(b*T+t)][D] bf16 (LDS 32x32 tile)
// so GEMM A-fragments become contiguous 16B loads.
// ---------------------------------------------------------------------------
__global__ __launch_bounds__(256, 1) void xcvt_kernel(
    const float* __restrict__ x, unsigned short* __restrict__ xbf)
{
    __shared__ unsigned short tile[32][34];         // padded vs bank conflicts
    const int tt = blockIdx.x * 32;                 // t tile
    const int k0 = blockIdx.y * 32;                 // k tile
    const int b  = blockIdx.z;
    const int c  = threadIdx.x & 31;
    const int r0 = threadIdx.x >> 5;

    const float* xp = x + ((size_t)b * DDIM + k0) * TDIM + tt;
    #pragma unroll
    for (int j = 0; j < 4; ++j) {
        int r = r0 + j * 8;
        tile[r][c] = f2bf(xp[(size_t)r * TDIM + c]);
    }
    __syncthreads();
    unsigned short* op = xbf + ((size_t)b * TDIM + tt) * DDIM + k0;
    #pragma unroll
    for (int j = 0; j < 4; ++j) {
        int t = r0 + j * 8;
        op[(size_t)t * DDIM + c] = tile[c][t];
    }
}

// ---------------------------------------------------------------------------
// Kernel 3: zraw = cos(x^T @ rff_w + rff_b) -> bf16 [BT][512] (pre-layernorm).
// CTA: 256 threads (8 waves), tile M=128 x N=64. Wave w: mg=w>>1, ng=w&1;
// wave tile 32x32 -> acc[2][2] = 32 VGPRs (fits the per-wave budget, no spill).
// B panel 64 cols x 32 K (4 KB) double-buffered in LDS via async copies.
// ---------------------------------------------------------------------------
__global__ __launch_bounds__(256, 1) void rff_gemm_kernel(
    const unsigned short* __restrict__ xbf, const unsigned short* __restrict__ wT,
    const float* __restrict__ rff_b, unsigned short* __restrict__ zbf)
{
    __shared__ alignas(16) unsigned short Bs[2][64 * BSTRIDE];   // 2 x 5 KB

    const int tid  = threadIdx.x;
    const int lane = tid & 31;
    const int w    = tid >> 5;
    const int mg = w >> 1, ng = w & 1;
    const int lm = lane & 15, lh = lane >> 4;
    const int rowbase  = blockIdx.x * 128;
    const int ncolbase = blockIdx.y * 64;

    const int scol = tid >> 2, scc = tid & 3;        // staging: 256 x 16B chunks
    const unsigned short* bsrc = wT + (size_t)(ncolbase + scol) * DDIM + scc * 8;

    f32x8 acc[2][2];
    #pragma unroll
    for (int mt = 0; mt < 2; ++mt)
        #pragma unroll
        for (int nt = 0; nt < 2; ++nt)
            #pragma unroll
            for (int e = 0; e < 8; ++e) acc[mt][nt][e] = 0.0f;

    // Prologue: stage k0 = 0 into buffer 0.
    async_copy16(&Bs[0][scol * BSTRIDE + scc * 8], bsrc);
    async_wait();
    __syncthreads();

    for (int k0 = 0; k0 < DDIM; k0 += 32) {
        const int cur = (k0 >> 5) & 1, nxt = cur ^ 1;
        if (k0 + 32 < DDIM)      // prefetch next panel while we compute
            async_copy16(&Bs[nxt][scol * BSTRIDE + scc * 8], bsrc + k0 + 32);

        Frag a[2];
        #pragma unroll
        for (int mt = 0; mt < 2; ++mt) {
            const u32x4* p = reinterpret_cast<const u32x4*>(
                xbf + (size_t)(rowbase + mg * 32 + mt * 16 + lm) * DDIM + k0 + lh * 8);
            a[mt].q[0] = p[0];
            a[mt].q[1] = p[2];                        // +16 shorts
        }
        #pragma unroll
        for (int nt = 0; nt < 2; ++nt) {
            Frag bfr;
            const u32x4* bp = reinterpret_cast<const u32x4*>(
                &Bs[cur][(ng * 32 + nt * 16 + lm) * BSTRIDE + lh * 8]);
            bfr.q[0] = bp[0];
            bfr.q[1] = bp[2];
            #pragma unroll
            for (int mt = 0; mt < 2; ++mt)
                acc[mt][nt] = __builtin_amdgcn_wmma_f32_16x16x32_bf16(
                    false, a[mt].v, false, bfr.v, (short)0, acc[mt][nt], false, false);
        }
        async_wait();
        __syncthreads();
    }

    // Epilogue: cos(acc + bias) -> bf16 (layernorm done by ln_kernel).
    #pragma unroll
    for (int nt = 0; nt < 2; ++nt) {
        int ncol = ncolbase + ng * 32 + nt * 16 + lm;
        float bv = rff_b[ncol];
        #pragma unroll
        for (int mt = 0; mt < 2; ++mt)
            #pragma unroll
            for (int v = 0; v < 8; ++v) {
                int row = rowbase + mg * 32 + mt * 16 + lh * 8 + v;
                zbf[(size_t)row * LDIM + ncol] = f2bf(cosf(acc[mt][nt][v] + bv));
            }
    }
}

// ---------------------------------------------------------------------------
// Kernel 4: in-place layernorm of z over the feature axis (512).
// One row per wave32: 16 bf16 per lane, __shfl_xor reduction.
// ---------------------------------------------------------------------------
__global__ __launch_bounds__(256, 1) void ln_kernel(unsigned short* __restrict__ zbf)
{
    const int lane = threadIdx.x & 31;
    const int w    = threadIdx.x >> 5;
    const size_t row = (size_t)blockIdx.x * 8 + w;

    unsigned short* rp = zbf + row * LDIM + lane * 16;
    Frag d;
    const u32x4* p = reinterpret_cast<const u32x4*>(rp);
    d.q[0] = p[0]; d.q[1] = p[1];

    float v[16], s = 0.0f, s2 = 0.0f;
    #pragma unroll
    for (int i = 0; i < 16; ++i) { v[i] = bf2f(d.us[i]); s += v[i]; s2 += v[i] * v[i]; }
    #pragma unroll
    for (int off = 16; off > 0; off >>= 1) {
        s  += __shfl_xor(s,  off, 32);
        s2 += __shfl_xor(s2, off, 32);
    }
    float mu  = s  * (1.0f / LDIM);
    float var = s2 * (1.0f / LDIM) - mu * mu;
    float rs  = rsqrtf(var + 1e-5f);
    #pragma unroll
    for (int i = 0; i < 16; ++i) d.us[i] = f2bf((v[i] - mu) * rs);

    u32x4* o = reinterpret_cast<u32x4*>(rp);
    o[0] = d.q[0]; o[1] = d.q[1];
}

// ---------------------------------------------------------------------------
// Kernel 5: gate = act(z @ W + z_prev @ R + bias) -> bf16 [3][BT][512].
// Grid: (BT/128, 8 n-blocks, 3 gates). CTA: 256 threads, tile M=128 x N=64.
// Wave tile 32x32, two B streams (W,R) double-buffered in LDS (4 x 5 KB).
// ---------------------------------------------------------------------------
__global__ __launch_bounds__(256, 1) void gates_kernel(
    const unsigned short* __restrict__ zbf, const unsigned short* __restrict__ wbf,
    const float* __restrict__ b_i, const float* __restrict__ b_f,
    const float* __restrict__ b_g, const float* __restrict__ init_latent,
    unsigned short* __restrict__ gout)
{
    __shared__ alignas(16) unsigned short Wsm[2][64 * BSTRIDE];
    __shared__ alignas(16) unsigned short Rsm[2][64 * BSTRIDE];

    const int tid  = threadIdx.x;
    const int lane = tid & 31;
    const int w    = tid >> 5;
    const int mg = w >> 1, ng = w & 1;
    const int lm = lane & 15, lh = lane >> 4;

    const int rowbase  = blockIdx.x * 128;
    const int ncolbase = blockIdx.y * 64;
    const int gate     = blockIdx.z;

    const unsigned short* W = wbf + (size_t)(1 + gate) * (LDIM * LDIM);
    const unsigned short* R = wbf + (size_t)(4 + gate) * (LDIM * LDIM);

    const int scol = tid >> 2, scc = tid & 3;
    const size_t goff0 = (size_t)(ncolbase + scol) * LDIM + scc * 8;

    f32x8 acc[2][2];
    #pragma unroll
    for (int mt = 0; mt < 2; ++mt)
        #pragma unroll
        for (int nt = 0; nt < 2; ++nt)
            #pragma unroll
            for (int e = 0; e < 8; ++e) acc[mt][nt][e] = 0.0f;

    // Prologue: stage k0 = 0.
    async_copy16(&Wsm[0][scol * BSTRIDE + scc * 8], W + goff0);
    async_copy16(&Rsm[0][scol * BSTRIDE + scc * 8], R + goff0);
    async_wait();
    __syncthreads();

    for (int k0 = 0; k0 < LDIM; k0 += 32) {
        const int cur = (k0 >> 5) & 1, nxt = cur ^ 1;
        if (k0 + 32 < LDIM) {
            async_copy16(&Wsm[nxt][scol * BSTRIDE + scc * 8], W + goff0 + k0 + 32);
            async_copy16(&Rsm[nxt][scol * BSTRIDE + scc * 8], R + goff0 + k0 + 32);
        }

        Frag az[2], ap[2];
        #pragma unroll
        for (int mt = 0; mt < 2; ++mt) {
            const int arow = rowbase + mg * 32 + mt * 16 + lm;
            const u32x4* pz = reinterpret_cast<const u32x4*>(
                zbf + (size_t)arow * LDIM + k0 + lh * 8);
            az[mt].q[0] = pz[0];
            az[mt].q[1] = pz[2];
            if ((arow & (TDIM - 1)) == 0) {           // t==0 -> init_latent
                #pragma unroll
                for (int i = 0; i < 8; ++i) {
                    ap[mt].us[i]     = f2bf(init_latent[k0 + lh * 8 + i]);
                    ap[mt].us[8 + i] = f2bf(init_latent[k0 + lh * 8 + 16 + i]);
                }
            } else {
                const u32x4* pp = reinterpret_cast<const u32x4*>(
                    zbf + (size_t)(arow - 1) * LDIM + k0 + lh * 8);
                ap[mt].q[0] = pp[0];
                ap[mt].q[1] = pp[2];
            }
        }

        #pragma unroll
        for (int nt = 0; nt < 2; ++nt) {
            Frag wf, rf;
            const u32x4* wp = reinterpret_cast<const u32x4*>(
                &Wsm[cur][(ng * 32 + nt * 16 + lm) * BSTRIDE + lh * 8]);
            wf.q[0] = wp[0]; wf.q[1] = wp[2];
            const u32x4* rp = reinterpret_cast<const u32x4*>(
                &Rsm[cur][(ng * 32 + nt * 16 + lm) * BSTRIDE + lh * 8]);
            rf.q[0] = rp[0]; rf.q[1] = rp[2];
            #pragma unroll
            for (int mt = 0; mt < 2; ++mt) {
                acc[mt][nt] = __builtin_amdgcn_wmma_f32_16x16x32_bf16(
                    false, az[mt].v, false, wf.v, (short)0, acc[mt][nt], false, false);
                acc[mt][nt] = __builtin_amdgcn_wmma_f32_16x16x32_bf16(
                    false, ap[mt].v, false, rf.v, (short)0, acc[mt][nt], false, false);
            }
        }
        async_wait();
        __syncthreads();
    }

    const float* bp = (gate == 0) ? b_i : (gate == 1) ? b_f : b_g;
    unsigned short* gdst = gout + (size_t)gate * BT * LDIM;

    #pragma unroll
    for (int nt = 0; nt < 2; ++nt) {
        int ncol = ncolbase + ng * 32 + nt * 16 + lm;
        float bv = bp[ncol];
        #pragma unroll
        for (int mt = 0; mt < 2; ++mt)
            #pragma unroll
            for (int v = 0; v < 8; ++v) {
                int row = rowbase + mg * 32 + mt * 16 + lh * 8 + v;
                float xv = acc[mt][nt][v] + bv;
                float val = (gate < 2) ? (1.0f / (1.0f + expf(-xv))) : tanhf(xv);
                gdst[(size_t)row * LDIM + ncol] = f2bf(val);
            }
    }
}

// ---------------------------------------------------------------------------
// Kernel 6: sequential scan c_t = f*c + i*g, h = sin(z)*c; output [B,L,T] f32.
// One thread per (b,l) chain; 16 t-steps buffered in registers so each thread
// writes a full 64B cacheline of contiguous t into the transposed output.
// ---------------------------------------------------------------------------
__global__ __launch_bounds__(256, 1) void scan_kernel(
    const unsigned short* __restrict__ zbf, const unsigned short* __restrict__ gates,
    const float* __restrict__ init_context, float* __restrict__ out)
{
    const int tid = threadIdx.x;
    const int b   = blockIdx.x >> 1;
    const int l   = ((blockIdx.x & 1) << 8) + tid;

    const size_t NG = (size_t)BT * LDIM;
    const unsigned short* gi = gates;
    const unsigned short* gf = gates + NG;
    const unsigned short* gg = gates + 2 * NG;

    float c = init_context[l];
    const size_t rbase = (size_t)b * TDIM;
    float* ob = out + (size_t)b * LDIM * TDIM + (size_t)l * TDIM;

    for (int t0 = 0; t0 < TDIM; t0 += 16) {
        float h[16];
        #pragma unroll
        for (int j = 0; j < 16; ++j) {
            size_t off = (rbase + t0 + j) * LDIM + l;
            float iv = bf2f(gi[off]);
            float fv = bf2f(gf[off]);
            float gv = bf2f(gg[off]);
            float ov = sinf(bf2f(zbf[off]));
            c = fv * c + iv * gv;
            h[j] = ov * c;
        }
        #pragma unroll
        for (int j = 0; j < 16; j += 4) {
            f32x4 v; v[0] = h[j]; v[1] = h[j + 1]; v[2] = h[j + 2]; v[3] = h[j + 3];
            *reinterpret_cast<f32x4*>(ob + t0 + j) = v;
        }
    }
}

// ---------------------------------------------------------------------------
extern "C" void kernel_launch(void* const* d_in, const int* in_sizes, int n_in,
                              void* d_out, int out_size, void* d_ws, size_t ws_size,
                              hipStream_t stream)
{
    const float* x           = (const float*)d_in[0];
    const float* rff_w       = (const float*)d_in[1];
    const float* rff_b       = (const float*)d_in[2];
    const float* w_i         = (const float*)d_in[3];
    const float* w_f         = (const float*)d_in[4];
    const float* w_g         = (const float*)d_in[5];
    const float* r_i         = (const float*)d_in[6];
    const float* r_f         = (const float*)d_in[7];
    const float* r_g         = (const float*)d_in[8];
    const float* b_i         = (const float*)d_in[9];
    const float* b_f         = (const float*)d_in[10];
    const float* b_g         = (const float*)d_in[11];
    const float* init_latent = (const float*)d_in[12];
    const float* init_ctx    = (const float*)d_in[13];
    float* out = (float*)d_out;

    // Workspace layout (bytes):
    //   zbf   : BT*L bf16              =  64 MB   @ 0      (raw, then LN in place)
    //   gates : 3 * BT*L bf16          = 192 MB   @ 64 MB
    //   wbf   : 7 * 512*512 bf16       = 3.5 MB   @ 256 MB
    //   xbf   : BT*D bf16              =  64 MB   @ 260 MB
    char* ws = (char*)d_ws;
    unsigned short* zbf   = (unsigned short*)ws;
    unsigned short* gates = (unsigned short*)(ws + ((size_t)64 << 20));
    unsigned short* wbf   = (unsigned short*)(ws + ((size_t)256 << 20));
    unsigned short* xbf   = (unsigned short*)(ws + ((size_t)260 << 20));

    wcvt_kernel<<<7 * 262144 / 256, 256, 0, stream>>>(
        rff_w, w_i, w_f, w_g, r_i, r_f, r_g, wbf);

    dim3 gx(TDIM / 32, DDIM / 32, BDIM);
    xcvt_kernel<<<gx, 256, 0, stream>>>(x, xbf);

    dim3 g1(BT / 128, LDIM / 64);
    rff_gemm_kernel<<<g1, 256, 0, stream>>>(xbf, wbf, rff_b, zbf);

    ln_kernel<<<BT / 8, 256, 0, stream>>>(zbf);

    dim3 g2(BT / 128, LDIM / 64, 3);
    gates_kernel<<<g2, 256, 0, stream>>>(zbf, wbf, b_i, b_f, b_g, init_latent, gates);

    scan_kernel<<<BDIM * 2, 256, 0, stream>>>(zbf, gates, init_ctx, out);
}
